// SceneGNN_71871982731329
// MI455X (gfx1250) — matrix operand order repaired and checked by reading, not compiled, not run on hardware
//
#include <hip/hip_runtime.h>
#include <hip/hip_bf16.h>

#define NN 100000
#define EE 1600000
#define GG 512
#define INF_ 128
#define HID 64
#define EPSF 1e-5f

typedef __attribute__((ext_vector_type(16))) _Float16 v16h;
typedef __attribute__((ext_vector_type(8)))  _Float16 v8h;
typedef __attribute__((ext_vector_type(8)))  float    v8f;

// ---------- helpers ----------
__device__ inline void atomAddF(float* p, float v) {
    // native global_atomic_add_f32; agg buffers are L2-resident (25.6MB << 192MB L2)
    unsafeAtomicAdd(p, v);
}

__device__ inline void atomMaxF(float* addr, float value) {
    // IEEE float ordering via int/uint atomics (works incl. -inf init)
    if (value >= 0.0f) {
        atomicMax((int*)addr, __float_as_int(value));
    } else {
        atomicMin((unsigned int*)addr, __float_as_uint(value));
    }
}

// ---------- init ----------
__global__ __launch_bounds__(256)
void init_kernel(float* aggB, float* aggC, float* deg,
                 float* pmean, float* pmax, float* cnt) {
    size_t idx = (size_t)blockIdx.x * 256 + threadIdx.x;
    if (idx < (size_t)NN * HID) { aggB[idx] = 0.0f; aggC[idx] = 0.0f; }
    if (idx < NN) deg[idx] = 1.0f;                 // self-loop
    if (idx < (size_t)GG * HID) { pmean[idx] = 0.0f; pmax[idx] = -INFINITY; }
    if (idx < GG) cnt[idx] = 0.0f;
}

// ---------- weight transpose to f16: wt[n*K+k] = (f16)W[k*64+n] ----------
__global__ __launch_bounds__(256)
void wtrans_kernel(const float* __restrict__ W, _Float16* __restrict__ wt, int K) {
    int idx = blockIdx.x * 256 + threadIdx.x;
    if (idx < K * HID) {
        int n = idx / K;
        int k = idx - n * K;
        wt[idx] = (_Float16)W[k * HID + n];
    }
}

// ---------- degree + dinv ----------
__global__ __launch_bounds__(256)
void degree_kernel(const long long* __restrict__ ei, float* __restrict__ deg) {
    int e = blockIdx.x * 256 + threadIdx.x;
    if (e < EE) atomAddF(&deg[(int)ei[EE + e]], 1.0f);
}

__global__ __launch_bounds__(256)
void rsqrt_kernel(float* d) {
    int i = blockIdx.x * 256 + threadIdx.x;
    if (i < NN) d[i] = rsqrtf(d[i]);
}

// ---------- WMMA GEMM:  C[rows,64] = A[rows,K] @ Wt^T   (f16 in, f32 acc) ----------
// BT is W pre-transposed to f16 [64][K] so the B fragment is two contiguous 8-half runs.
// 1 block = 4 waves; wave w owns the 16x16 tile (rows 16*blockIdx, cols 16*w).
// Fragment layouts per CDNA5 ISA 7.12.2 (wave32). K templated -> full unroll.
template <int K>
__global__ __launch_bounds__(128)
void gemm_wmma_kernel(const float* __restrict__ A, const _Float16* __restrict__ BT,
                      float* __restrict__ C) {
    const int wave = threadIdx.x >> 5;
    const int lane = threadIdx.x & 31;
    const int m    = lane & 15;
    const int hi   = lane >> 4;          // lane half selects K-group
    const int row0 = blockIdx.x * 16;
    const int col0 = wave * 16;

    const float*    arow = A  + (size_t)(row0 + m) * K;
    const _Float16* brow = BT + (size_t)(col0 + m) * K + hi * 8;

    v8f c = {};
    #pragma unroll
    for (int kt = 0; kt < K; kt += 32) {
        // A 16x32 f16 fragment: two contiguous 8-float runs per lane -> 4x b128
        const float4 r0 = *(const float4*)(arow + kt + hi * 8);
        const float4 r1 = *(const float4*)(arow + kt + hi * 8 + 4);
        const float4 r2 = *(const float4*)(arow + kt + 16 + hi * 8);
        const float4 r3 = *(const float4*)(arow + kt + 16 + hi * 8 + 4);
        v16h a;
        a[0]  = (_Float16)r0.x; a[1]  = (_Float16)r0.y;
        a[2]  = (_Float16)r0.z; a[3]  = (_Float16)r0.w;
        a[4]  = (_Float16)r1.x; a[5]  = (_Float16)r1.y;
        a[6]  = (_Float16)r1.z; a[7]  = (_Float16)r1.w;
        a[8]  = (_Float16)r2.x; a[9]  = (_Float16)r2.y;
        a[10] = (_Float16)r2.z; a[11] = (_Float16)r2.w;
        a[12] = (_Float16)r3.x; a[13] = (_Float16)r3.y;
        a[14] = (_Float16)r3.z; a[15] = (_Float16)r3.w;

        // B 32x16 f16 fragment: two contiguous 8-half runs per lane -> 2x b128
        const v8h b0 = *(const v8h*)(brow + kt);
        const v8h b1 = *(const v8h*)(brow + kt + 16);
        v16h b;
        #pragma unroll
        for (int e = 0; e < 8; ++e) { b[e] = b0[e]; b[8 + e] = b1[e]; }

        c = __builtin_amdgcn_wmma_f32_16x16x32_f16(
                false, a, false, b, (short)0, c, false, false);
    }
    // C/D: VGPR r -> M = hi*8 + r, N = m
    float* crow = C + (size_t)(row0 + hi * 8) * HID + col0 + m;
    #pragma unroll
    for (int r = 0; r < 8; ++r) crow[(size_t)r * HID] = c[r];
}

// ---------- edge scatter: agg[dst] += hw[src] * dinv[src]*dinv[dst] ----------
// 16 edges per block; (src,dst,norm) staged once in LDS, then 16 lanes x float4
// gathers + native f32 atomics into L2.
__global__ __launch_bounds__(256)
void scatter_kernel(const long long* __restrict__ ei,
                    const float* __restrict__ dinv,
                    const float* __restrict__ hw,
                    float* __restrict__ agg) {
    __shared__ int   lsrc[16];
    __shared__ int   ldst[16];
    __shared__ float lnrm[16];
    const int t  = threadIdx.x;
    const int e0 = blockIdx.x * 16;
    if (t < 16) {
        int s = (int)ei[e0 + t];
        int d = (int)ei[EE + e0 + t];
        lsrc[t] = s;
        ldst[t] = d;
        lnrm[t] = dinv[s] * dinv[d];
    }
    __syncthreads();
    const int el = t >> 4;
    const int f4 = (t & 15) * 4;
    const int s = lsrc[el];
    const int d = ldst[el];
    const float nrm = lnrm[el];
    const float4 v = *reinterpret_cast<const float4*>(hw + (size_t)s * HID + f4);
    float* out = agg + (size_t)d * HID + f4;
    atomAddF(out + 0, v.x * nrm);
    atomAddF(out + 1, v.y * nrm);
    atomAddF(out + 2, v.z * nrm);
    atomAddF(out + 3, v.w * nrm);
}

// ---------- fused self-loop + bias + BN + ReLU (+ residual), float4, in place ----------
__global__ __launch_bounds__(256)
void epilogue_kernel(const float* __restrict__ hw, const float* __restrict__ dinv,
                     const float* __restrict__ b,  const float* __restrict__ g,
                     const float* __restrict__ be, const float* __restrict__ rm,
                     const float* __restrict__ rv, const float* __restrict__ res,
                     float* __restrict__ agg) {
    int idx = blockIdx.x * 256 + threadIdx.x;            // over NN*16
    if (idx >= NN * 16) return;
    const int i  = idx >> 4;
    const int f4 = (idx & 15) * 4;
    const size_t base = (size_t)i * HID + f4;
    const float di2 = dinv[i] * dinv[i];

    const float4 hv = *(const float4*)(hw + base);
    const float4 av = *(const float4*)(agg + base);
    const float4 bv = *(const float4*)(b + f4);
    const float4 gv = *(const float4*)(g + f4);
    const float4 ev = *(const float4*)(be + f4);
    const float4 mv = *(const float4*)(rm + f4);
    const float4 vv = *(const float4*)(rv + f4);

    float4 o;
    o.x = fmaxf((av.x + hv.x * di2 + bv.x - mv.x) * rsqrtf(vv.x + EPSF) * gv.x + ev.x, 0.0f);
    o.y = fmaxf((av.y + hv.y * di2 + bv.y - mv.y) * rsqrtf(vv.y + EPSF) * gv.y + ev.y, 0.0f);
    o.z = fmaxf((av.z + hv.z * di2 + bv.z - mv.z) * rsqrtf(vv.z + EPSF) * gv.z + ev.z, 0.0f);
    o.w = fmaxf((av.w + hv.w * di2 + bv.w - mv.w) * rsqrtf(vv.w + EPSF) * gv.w + ev.w, 0.0f);
    if (res) {
        const float4 rr = *(const float4*)(res + base);
        o.x += rr.x; o.y += rr.y; o.z += rr.z; o.w += rr.w;
    }
    *(float4*)(agg + base) = o;
}

// ---------- per-graph mean/max pooling ----------
__global__ __launch_bounds__(256)
void pool_kernel(const float* __restrict__ h, const long long* __restrict__ batch,
                 float* __restrict__ pmean, float* __restrict__ pmax,
                 float* __restrict__ cnt) {
    int t = blockIdx.x * 256 + threadIdx.x;
    int i = t >> 6;
    int f = t & 63;
    if (i >= NN) return;
    int g = (int)batch[i];
    float v = h[(size_t)i * HID + f];
    atomAddF(&pmean[g * HID + f], v);
    atomMaxF(&pmax[g * HID + f], v);
    if (f == 0) atomAddF(&cnt[g], 1.0f);
}

// ---------- pooling finalize: hg[G,128] = [mean/cnt | max] ----------
__global__ __launch_bounds__(256)
void pool_finalize_kernel(const float* __restrict__ pmean, const float* __restrict__ pmax,
                          const float* __restrict__ cnt, float* __restrict__ hg) {
    int idx = blockIdx.x * 256 + threadIdx.x;            // over GG*128
    if (idx >= GG * 2 * HID) return;
    int g = idx >> 7;
    int f = idx & 127;
    hg[idx] = (f < HID) ? pmean[g * HID + f] / fmaxf(cnt[g], 1.0f)
                        : pmax[g * HID + (f - HID)];
}

// ---------- MLP head tail: relu(z1pre+bh1) -> 32 -> 1 -> sigmoid ----------
__global__ __launch_bounds__(64)
void head_tail_kernel(const float* __restrict__ z1pre, const float* __restrict__ bh1,
                      const float* __restrict__ Wh2,   const float* __restrict__ bh2,
                      const float* __restrict__ Wh3,   const float* __restrict__ bh3,
                      float* __restrict__ out) {
    __shared__ float z1[HID];
    __shared__ float z2[HID / 2];
    const int g = blockIdx.x;
    const int t = threadIdx.x;

    z1[t] = fmaxf(z1pre[g * HID + t] + bh1[t], 0.0f);
    __syncthreads();

    if (t < HID / 2) {
        float a2 = bh2[t];
        #pragma unroll 4
        for (int k = 0; k < HID; ++k) a2 += z1[k] * Wh2[k * (HID / 2) + t];
        z2[t] = fmaxf(a2, 0.0f);
    }
    __syncthreads();

    if (t == 0) {
        float a3 = bh3[0];
        #pragma unroll
        for (int k = 0; k < HID / 2; ++k) a3 += z2[k] * Wh3[k];
        out[g] = 1.0f / (1.0f + expf(-a3));
    }
}

// ---------- launch ----------
extern "C" void kernel_launch(void* const* d_in, const int* in_sizes, int n_in,
                              void* d_out, int out_size, void* d_ws, size_t ws_size,
                              hipStream_t stream) {
    const float*     x     = (const float*)d_in[0];
    const long long* ei    = (const long long*)d_in[1];
    const long long* batch = (const long long*)d_in[2];
    const float* W1  = (const float*)d_in[3];
    const float* b1  = (const float*)d_in[4];
    const float* g1  = (const float*)d_in[5];
    const float* be1 = (const float*)d_in[6];
    const float* rm1 = (const float*)d_in[7];
    const float* rv1 = (const float*)d_in[8];
    const float* W2  = (const float*)d_in[9];
    const float* b2  = (const float*)d_in[10];
    const float* g2  = (const float*)d_in[11];
    const float* be2 = (const float*)d_in[12];
    const float* rm2 = (const float*)d_in[13];
    const float* rv2 = (const float*)d_in[14];
    const float* Wh1 = (const float*)d_in[15];
    const float* bh1 = (const float*)d_in[16];
    const float* Wh2 = (const float*)d_in[17];
    const float* bh2 = (const float*)d_in[18];
    const float* Wh3 = (const float*)d_in[19];
    const float* bh3 = (const float*)d_in[20];
    float* out = (float*)d_out;

    float* ws    = (float*)d_ws;
    float* deg   = ws;                                 // N       (deg -> dinv in place)
    float* bufA  = deg   + NN;                         // N*64    (hw1, then hw2)
    float* bufB  = bufA  + (size_t)NN * HID;           // N*64    (agg1 -> h1)
    float* bufC  = bufB  + (size_t)NN * HID;           // N*64    (agg2 -> h2)
    float* pmean = bufC  + (size_t)NN * HID;           // G*64
    float* pmax  = pmean + (size_t)GG * HID;           // G*64
    float* cnt   = pmax  + (size_t)GG * HID;           // G
    float* hg    = cnt   + GG;                         // G*128
    float* z1p   = hg    + (size_t)GG * 2 * HID;       // G*64
    _Float16* wt1 = (_Float16*)(z1p + (size_t)GG * HID);   // 128*64 f16 (16B aligned)
    _Float16* wt2 = wt1 + INF_ * HID;                      // 64*64  f16
    _Float16* wth = wt2 + HID * HID;                       // 128*64 f16

    const int nhid_blocks = (NN * HID + 255) / 256;    // 25000

    init_kernel<<<nhid_blocks, 256, 0, stream>>>(bufB, bufC, deg, pmean, pmax, cnt);
    wtrans_kernel<<<(INF_ * HID + 255) / 256, 256, 0, stream>>>(W1, wt1, INF_);
    wtrans_kernel<<<(HID * HID + 255) / 256, 256, 0, stream>>>(W2, wt2, HID);
    wtrans_kernel<<<(INF_ * HID + 255) / 256, 256, 0, stream>>>(Wh1, wth, INF_);
    degree_kernel<<<(EE + 255) / 256, 256, 0, stream>>>(ei, deg);
    rsqrt_kernel<<<(NN + 255) / 256, 256, 0, stream>>>(deg);

    // layer 1
    gemm_wmma_kernel<INF_><<<NN / 16, 128, 0, stream>>>(x, wt1, bufA);
    scatter_kernel<<<EE / 16, 256, 0, stream>>>(ei, deg, bufA, bufB);
    epilogue_kernel<<<(NN * 16 + 255) / 256, 256, 0, stream>>>(bufA, deg, b1, g1, be1,
                                                               rm1, rv1, nullptr, bufB);
    // layer 2 (+residual)
    gemm_wmma_kernel<HID><<<NN / 16, 128, 0, stream>>>(bufB, wt2, bufA);
    scatter_kernel<<<EE / 16, 256, 0, stream>>>(ei, deg, bufA, bufC);
    epilogue_kernel<<<(NN * 16 + 255) / 256, 256, 0, stream>>>(bufA, deg, b2, g2, be2,
                                                               rm2, rv2, bufB, bufC);
    // pooling + head (first head GEMM also via WMMA)
    pool_kernel<<<nhid_blocks, 256, 0, stream>>>(bufC, batch, pmean, pmax, cnt);
    pool_finalize_kernel<<<(GG * 2 * HID + 255) / 256, 256, 0, stream>>>(pmean, pmax, cnt, hg);
    gemm_wmma_kernel<INF_><<<GG / 16, 128, 0, stream>>>(hg, wth, z1p);
    head_tail_kernel<<<GG, 64, 0, stream>>>(z1p, bh1, Wh2, bh2, Wh3, bh3, out);
}
